// LSTM_37271726195401
// MI455X (gfx1250) — matrix operand order repaired and checked by reading, not compiled, not run on hardware
//
#include <hip/hip_runtime.h>
#include <hip/hip_bf16.h>
#include <math.h>

#define B_ 64
#define T_ 256
#define F_ 512
#define H_ 512
#define G_ 2048   // 4*H

#define LDSPITCH 520    // 512 + 8 ushort pad -> 16-byte bank step for b128 loads
#define ZPITCH   18     // f32 z-tile row pitch: lanes 0-15 / 16-31 hit disjoint banks

typedef __attribute__((ext_vector_type(16))) __bf16 v16bf;
typedef __attribute__((ext_vector_type(8)))  float  v8f;

union ABPack {
    v16bf v;
    unsigned short s[16];
    uint4 q[2];
};

// round-to-nearest-even f32 -> bf16 bits
__device__ __forceinline__ unsigned short f2bf(float f) {
    union { float f; unsigned u; } x; x.f = f;
    unsigned r = x.u + 0x7FFFu + ((x.u >> 16) & 1u);
    return (unsigned short)(r >> 16);
}

__device__ __forceinline__ float sigmoidf_(float x) {
    return 1.0f / (1.0f + __expf(-x));
}

#define WMMA_BF16(A, Bm, C) \
    __builtin_amdgcn_wmma_f32_16x16x32_bf16(false, (A), false, (Bm), (short)0, (C), false, false)

// ---------------------------------------------------------------------------
// Kernel 1: convert X -> bf16, convert+transpose Wi/Wh -> bf16 [G][K],
// zero h double-buffer and barrier counters. Runs every launch.
// ---------------------------------------------------------------------------
__global__ void lstm_prep(const float* __restrict__ X,
                          const float* __restrict__ Wi, const float* __restrict__ Wh,
                          unsigned short* __restrict__ Xbf,
                          unsigned short* __restrict__ WiT, unsigned short* __restrict__ WhT,
                          unsigned short* __restrict__ hbuf, int* __restrict__ bar) {
    const int stride = gridDim.x * blockDim.x;
    const int tid0 = blockIdx.x * blockDim.x + threadIdx.x;

    for (int e = tid0; e < B_ * T_ * F_; e += stride) {
        Xbf[e] = f2bf(X[e]);
    }
    for (int e = tid0; e < G_ * F_; e += stride) {
        const int g = e >> 9;          // /F_
        const int k = e & (F_ - 1);
        WiT[e] = f2bf(Wi[(size_t)k * G_ + g]);
        WhT[e] = f2bf(Wh[(size_t)k * G_ + g]);
    }
    for (int e = tid0; e < 2 * B_ * H_; e += stride) {   // both h parity buffers
        hbuf[e] = 0;
    }
    for (int e = tid0; e < T_; e += stride) {
        bar[e] = 0;
    }
}

// ---------------------------------------------------------------------------
// Kernel 2: Zx = Xbf @ Wi + bias.  Block: 8 waves sharing one M-tile (A strip
// staged in LDS once), each wave does 4 N-tiles with one A fragment.
// ---------------------------------------------------------------------------
__global__ void __launch_bounds__(256) zx_gemm(const unsigned short* __restrict__ Xbf,
                                               const unsigned short* __restrict__ WiT,
                                               const float* __restrict__ bias,
                                               float* __restrict__ Zx) {
    __shared__ unsigned short ldsx[16 * LDSPITCH];

    const int wave = threadIdx.x >> 5;
    const int lane = threadIdx.x & 31;
    const int tm = blockIdx.x >> 2;                       // 0..1023 (shared by block)
    const int tg = (blockIdx.x & 3) * 8 + wave;           // 0..31: cols [tg*64, tg*64+64)
    const int lo = lane & 15;
    const bool hi = lane >= 16;

    for (int idx = threadIdx.x; idx < 16 * 64; idx += 256) {   // 16 rows x 64 uint4
        const int r = idx >> 6;
        const int j = idx & 63;
        *((uint4*)(ldsx + r * LDSPITCH) + j) =
            *((const uint4*)(Xbf + (size_t)(tm * 16 + r) * F_) + j);
    }
    __syncthreads();

    const unsigned short* arow = ldsx + lo * LDSPITCH;
    const unsigned short* wrow = WiT + (size_t)(tg * 64 + lo) * F_;

    v8f acc0 = {}, acc1 = {}, acc2 = {}, acc3 = {};
    for (int k0 = 0; k0 < F_; k0 += 32) {
        const int ka = k0 + (hi ? 8 : 0);
        ABPack a;
        a.q[0] = *(const uint4*)(arow + ka);
        a.q[1] = *(const uint4*)(arow + ka + 16);
        const int kb = k0 + (hi ? 8 : 0);
        ABPack b0, b1, b2, b3;
        b0.q[0] = *(const uint4*)(wrow + kb);
        b0.q[1] = *(const uint4*)(wrow + kb + 16);
        b1.q[0] = *(const uint4*)(wrow + 16 * F_ + kb);
        b1.q[1] = *(const uint4*)(wrow + 16 * F_ + kb + 16);
        b2.q[0] = *(const uint4*)(wrow + 32 * F_ + kb);
        b2.q[1] = *(const uint4*)(wrow + 32 * F_ + kb + 16);
        b3.q[0] = *(const uint4*)(wrow + 48 * F_ + kb);
        b3.q[1] = *(const uint4*)(wrow + 48 * F_ + kb + 16);
        acc0 = WMMA_BF16(a.v, b0.v, acc0);
        acc1 = WMMA_BF16(a.v, b1.v, acc1);
        acc2 = WMMA_BF16(a.v, b2.v, acc2);
        acc3 = WMMA_BF16(a.v, b3.v, acc3);
    }

    const int rbase = tm * 16 + (hi ? 8 : 0);
    v8f* accs[4] = { &acc0, &acc1, &acc2, &acc3 };
    for (int s = 0; s < 4; ++s) {
        const int col = tg * 64 + s * 16 + lo;
        const float bv = bias[col];
        for (int r = 0; r < 8; ++r) {
            Zx[(size_t)(rbase + r) * G_ + col] = (*accs[s])[r] + bv;
        }
    }
}

// ---------------------------------------------------------------------------
// Grid-wide barrier: one-shot counter per phase (no residual state).
// ---------------------------------------------------------------------------
__device__ __forceinline__ void grid_barrier(int* ctr, int nblk) {
    __threadfence();
    __syncthreads();
    if (threadIdx.x == 0) {
        atomicAdd(ctr, 1);
        while (__hip_atomic_load(ctr, __ATOMIC_ACQUIRE, __HIP_MEMORY_SCOPE_AGENT) < nblk) {
            __builtin_amdgcn_s_sleep(1);
        }
    }
    __syncthreads();
}

// ---------------------------------------------------------------------------
// Kernel 3: persistent recurrence, gate-grouped blocking.
// Block b owns h-columns [16b, 16b+16): its 8 waves (4 batch-tiles x 2
// gate-pairs) compute the i,f,g,o tiles for those columns. Gate exchange is
// block-local LDS (no global zbuf, no extra grid barrier); cell state c lives
// in LDS for the whole kernel. h is double-buffered in global so ONE grid
// barrier per step suffices. Wh slice (4 gates x 16 cols x 512 K bf16 = 64KB)
// is staged in LDS once and served by ds_load_b128 for all 256 steps.
// ---------------------------------------------------------------------------
__global__ void __launch_bounds__(256, 1) lstm_rec(const float* __restrict__ Zx,
                                                   const unsigned short* __restrict__ WhT,
                                                   const unsigned char* __restrict__ mask,
                                                   unsigned short* __restrict__ hbuf,
                                                   float* __restrict__ out,
                                                   int* __restrict__ bar,
                                                   int nblocks) {
    extern __shared__ char lds[];
    unsigned short* wcols = (unsigned short*)lds;                       // 64 x LDSPITCH bf16
    float* zls = (float*)(lds + (size_t)64 * LDSPITCH * 2);             // [4 gates][64 rows][ZPITCH]
    float* cls = zls + 4 * 64 * ZPITCH;                                  // [64 rows][16 cols]

    const int wave = threadIdx.x >> 5;
    const int lane = threadIdx.x & 31;
    const int tm = wave & 3;          // batch tile 0..3
    const int p  = wave >> 2;         // gate pair: 0 -> (i,f), 1 -> (g,o)
    const int lo = lane & 15;
    const bool hi = lane >= 16;
    const int C0 = blockIdx.x * 16;   // this block's h columns

    // ---- one-time: stage Wh columns {g*512 + C0 + c} (g=0..3, c=0..15) ----
    for (int idx = threadIdx.x; idx < 64 * 64; idx += 256) {   // 64 LDS cols x 64 uint4
        const int c64 = idx >> 6;
        const int j = idx & 63;
        const int g = c64 >> 4;
        const int c = c64 & 15;
        *((uint4*)(wcols + c64 * LDSPITCH) + j) =
            *((const uint4*)(WhT + (size_t)(g * 512 + C0 + c) * H_) + j);
    }
    for (int e = threadIdx.x; e < 64 * 16; e += 256) {
        cls[e] = 0.0f;                 // c0 = 0
    }
    __syncthreads();

    const int g0 = 2 * p, g1 = 2 * p + 1;
    const unsigned short* wl0 = wcols + (size_t)(g0 * 16 + lo) * LDSPITCH;
    const unsigned short* wl1 = wcols + (size_t)(g1 * 16 + lo) * LDSPITCH;
    const int colA = g0 * 512 + C0 + lo;      // Zx column for acc0
    const int colB = g1 * 512 + C0 + lo;      // Zx column for acc1
    const int rbase = tm * 16 + (hi ? 8 : 0);

    for (int t = 0; t < T_; ++t) {
        const unsigned short* hread  = hbuf + (size_t)(t & 1) * (B_ * H_);
        unsigned short*       hwrite = hbuf + (size_t)((t & 1) ^ 1) * (B_ * H_);

        // ---- init accumulators from precomputed Zx(t), prefetch Zx(t+1) ----
        v8f acc0, acc1;
        for (int r = 0; r < 8; ++r) {
            const size_t zrow = ((size_t)(rbase + r) * T_ + t) * G_;
            acc0[r] = Zx[zrow + colA];
            acc1[r] = Zx[zrow + colB];
            if (t + 1 < T_) {
                __builtin_prefetch(&Zx[zrow + G_ + colA], 0, 1);
            }
        }
        // ---- z tiles += h @ Wh  (A: global h, B: LDS) ----
        const unsigned short* hrow = hread + (size_t)(tm * 16 + lo) * H_;
        for (int k0 = 0; k0 < H_; k0 += 32) {
            const int ka = k0 + (hi ? 8 : 0);
            ABPack a;
            a.q[0] = *(const uint4*)(hrow + ka);
            a.q[1] = *(const uint4*)(hrow + ka + 16);
            const int kb = k0 + (hi ? 8 : 0);
            ABPack b0, b1;
            b0.q[0] = *(const uint4*)(wl0 + kb);
            b0.q[1] = *(const uint4*)(wl0 + kb + 16);
            b1.q[0] = *(const uint4*)(wl1 + kb);
            b1.q[1] = *(const uint4*)(wl1 + kb + 16);
            acc0 = WMMA_BF16(a.v, b0.v, acc0);
            acc1 = WMMA_BF16(a.v, b1.v, acc1);
        }
        // ---- publish gate tiles block-locally ----
        for (int r = 0; r < 8; ++r) {
            const int row = rbase + r;
            zls[(g0 * 64 + row) * ZPITCH + lo] = acc0[r];
            zls[(g1 * 64 + row) * ZPITCH + lo] = acc1[r];
        }
        __syncthreads();

        // ---- cell update: all 256 threads, 4 elements each ----
        for (int e = threadIdx.x; e < 64 * 16; e += 256) {
            const int b = e >> 4;
            const int j = e & 15;
            const float zi = zls[(0 * 64 + b) * ZPITCH + j];
            const float zf = zls[(1 * 64 + b) * ZPITCH + j];
            const float zg = zls[(2 * 64 + b) * ZPITCH + j];
            const float zo = zls[(3 * 64 + b) * ZPITCH + j];
            const float c  = cls[e];

            const float ig = sigmoidf_(zi);
            const float fg = sigmoidf_(zf);
            const float gg = tanhf(zg);
            const float og = sigmoidf_(zo);
            float cn = fg * c + ig * gg;
            float hn = og * tanhf(cn);
            const bool m = mask[b * T_ + t] != 0;
            cn = m ? cn : 0.0f;
            hn = m ? hn : 0.0f;

            cls[e] = cn;
            hwrite[(size_t)b * H_ + C0 + j] = f2bf(hn);
            out[((size_t)b * T_ + t) * H_ + C0 + j] = hn;
        }

        grid_barrier(&bar[t], nblocks);   // publishes h(t+1); also fences LDS reuse
    }
}

// ---------------------------------------------------------------------------
extern "C" void kernel_launch(void* const* d_in, const int* in_sizes, int n_in,
                              void* d_out, int out_size, void* d_ws, size_t ws_size,
                              hipStream_t stream) {
    const float*         X    = (const float*)d_in[0];          // [B,T,F]
    const unsigned char* mask = (const unsigned char*)d_in[1];  // [B,T] bool
    const float*         Wi   = (const float*)d_in[2];          // [F,4H]
    const float*         Wh   = (const float*)d_in[3];          // [H,4H]
    const float*         bias = (const float*)d_in[4];          // [4H]
    float*               out  = (float*)d_out;                  // [B,T,H]

    char* ws = (char*)d_ws;
    unsigned short* WiT  = (unsigned short*)ws;  ws += (size_t)G_ * F_ * sizeof(unsigned short);
    unsigned short* WhT  = (unsigned short*)ws;  ws += (size_t)G_ * H_ * sizeof(unsigned short);
    unsigned short* Xbf  = (unsigned short*)ws;  ws += (size_t)B_ * T_ * F_ * sizeof(unsigned short);
    unsigned short* hbuf = (unsigned short*)ws;  ws += (size_t)2 * B_ * H_ * sizeof(unsigned short);
    int* bar = (int*)ws;                         ws += (size_t)T_ * sizeof(int);
    ws = (char*)(((size_t)ws + 255) & ~(size_t)255);
    float* Zx = (float*)ws;                      // [B*T, 4H] f32 (~128 MB)

    const size_t lds_bytes = (size_t)64 * LDSPITCH * sizeof(unsigned short)   // Wh slice
                           + (size_t)4 * 64 * ZPITCH * sizeof(float)          // z exchange
                           + (size_t)64 * 16 * sizeof(float);                 // c state

    lstm_prep<<<2048, 256, 0, stream>>>(X, Wi, Wh, Xbf, WiT, WhT, hbuf, bar);
    zx_gemm<<<4096, 256, 0, stream>>>(Xbf, WiT, bias, Zx);
    lstm_rec<<<32, 256, lds_bytes, stream>>>(Zx, WhT, mask, hbuf, out, bar, 32);
}
